// EdgeGCN_70128226009697
// MI455X (gfx1250) — compile-verified, matrix operand order checked by reading, not run on hardware
//
#include <hip/hip_runtime.h>
#include <hip/hip_bf16.h>

typedef __attribute__((ext_vector_type(16))) __bf16 v16bf;
typedef __attribute__((ext_vector_type(8)))  float  v8f;

union FragA { v16bf v; __bf16 e[16]; unsigned int u[8]; uint4 q[2]; };
union FragC { v8f   v; float  e[8];  };

// k index of element h for 16-bit A/B WMMA fragments (ISA 7.12.2):
// lanes 0-15: h0..7 -> K0..7, h8..15 -> K16..23; lanes 16-31: +8.
__device__ __forceinline__ int akmap(int lane, int h) {
    return h + ((h >= 8) ? 8 : 0) + ((lane >= 16) ? 8 : 0);
}

__device__ __forceinline__ v8f wmma_bf16(v16bf a, v16bf b, v8f c) {
    return __builtin_amdgcn_wmma_f32_16x16x32_bf16(false, a, false, b, (short)0, c, false, false);
}

// Fragment loaders for a CONTIGUOUS 32-element K-window starting at p0
// (k pattern: lane<16 -> k 0..7 & 16..23 ; lane>=16 -> +8).
__device__ __forceinline__ void frag_ld_b128(FragA& f, const __bf16* p0, int lane) {
    const __bf16* p = p0 + ((lane >= 16) ? 8 : 0);        // 16B-aligned
    f.q[0] = *(const uint4*)(p);
    f.q[1] = *(const uint4*)(p + 16);
}
__device__ __forceinline__ void frag_ld_b32(FragA& f, const __bf16* p0, int lane) {
    const __bf16* p = p0 + ((lane >= 16) ? 8 : 0);        // 4B-aligned
#pragma unroll
    for (int j = 0; j < 4; ++j) {
        f.u[j]     = *(const unsigned int*)(p + 2 * j);
        f.u[4 + j] = *(const unsigned int*)(p + 16 + 2 * j);
    }
}

// ---------------------------------------------------------------------------
// Weight prep (fp32 -> padded, K-permuted bf16):
//  w1b [32][16] : k' = dk*6+ci  (<30, zero pad)   from conv1_w [16][6][5]
//  w2b [96][32] : k' = dk*16+ci (<80, zero pad)   from conv2_w [32][16][5]
//  fcb [128][3840] : k' = l*32+co                 from fc_w    [128][co*120+l]
//  g1b/g2b [128][128] : direct                    gcn1_w / gcn2_w
// ---------------------------------------------------------------------------
__global__ void prep_weights(const float* __restrict__ c1w, const float* __restrict__ c2w,
                             const float* __restrict__ fcw, const float* __restrict__ g1,
                             const float* __restrict__ g2,
                             __bf16* w1b, __bf16* w2b, __bf16* fcb, __bf16* g1b, __bf16* g2b) {
    int i = blockIdx.x * 256 + threadIdx.x;
    if (i < 512) {
        int k = i >> 4, co = i & 15;
        float v = 0.0f;
        if (k < 30) { int dk = k / 6, ci = k - dk * 6; v = c1w[co * 30 + ci * 5 + dk]; }
        w1b[i] = (__bf16)v;
    } else if (i < 512 + 3072) {
        int j = i - 512; int k = j >> 5, co = j & 31;
        float v = 0.0f;
        if (k < 80) { int dk = k >> 4, ci = k & 15; v = c2w[co * 80 + ci * 5 + dk]; }
        w2b[j] = (__bf16)v;
    } else if (i < 3584 + 491520) {
        int j = i - 3584;
        int o = j / 3840, r = j - o * 3840;
        int l = r >> 5, co = r & 31;
        fcb[j] = (__bf16)fcw[o * 3840 + co * 120 + l];
    } else if (i < 495104 + 16384) {
        int j = i - 495104; g1b[j] = (__bf16)g1[j];
    } else if (i < 511488 + 16384) {
        int j = i - 511488; g2b[j] = (__bf16)g2[j];
    }
}

// ---------------------------------------------------------------------------
// Fused conv1(6->16,k5,p2)+ReLU -> conv2(16->32,k5,p2)+ReLU -> fc(3840->128)
// One workgroup (256 thr = 8 wave32) per 16-node tile, whole tile in LDS.
// GEMM row mapping: row = l*16 + node  (no div/mod anywhere).
// LDS: region0 [0,122880): phase0/1 in2 tile [16][124][6]; phase2/3 y2 tile
//      region1 [122880,186368): y1 tile [16][124][16] (+-2 halo) + slack
// ---------------------------------------------------------------------------
#define FUSED_LDS (186368 + 128)
__global__ __launch_bounds__(256) void fused_conv_fc(
        const float* __restrict__ edge_attr,
        const __bf16* __restrict__ w1b, const __bf16* __restrict__ w2b,
        const __bf16* __restrict__ fcb,
        const float* __restrict__ c1bias, const float* __restrict__ c2bias,
        const float* __restrict__ fcbias,
        float* __restrict__ x0, int nN) {
    extern __shared__ char smem[];
    __bf16* in2    = (__bf16*)smem;             // [16][124][6]  node*744 + (l+2)*6 + ci
    __bf16* y2_lds = (__bf16*)smem;             // [16][120][32] node*3840 + l*32 + co
    __bf16* y1_lds = (__bf16*)(smem + 122880);  // [16][124][16] node*1984 + (l+2)*16 + ci

    const int tid  = threadIdx.x;
    const int lane = tid & 31;
    const int wv   = tid >> 5;
    const int col  = lane & 15;
    const int node = lane & 15;
    const int tile0 = blockIdx.x * 16;

    // ---- phase 0: stage input (coalesced global reads, transposed LDS store)
    for (int j = tid; j < 16 * 6 * 120; j += 256) {      // j = node*720 + ci*120 + l
        int nd = j / 720, rem = j - nd * 720;
        int ci = rem / 120, l = rem - ci * 120;
        float v = (tile0 + nd < nN) ? edge_attr[(size_t)tile0 * 720 + j] : 0.0f;
        in2[nd * 744 + (l + 2) * 6 + ci] = (__bf16)v;
    }
    // zero conv halos (l+2 in {0,1,122,123})
    for (int i = tid; i < 384; i += 256) {               // 16 * 4 * 6
        int nd = i / 24, rem = i - nd * 24;
        int lp4 = rem / 6, ci = rem - lp4 * 6;
        int lp = (lp4 < 2) ? lp4 : 120 + lp4;
        in2[nd * 744 + lp * 6 + ci] = (__bf16)0.0f;
    }
    for (int i = tid; i < 1024; i += 256) {              // 16 * 4 * 16
        int nd = i >> 6, rem = i & 63;
        int lp4 = rem >> 4, ci = rem & 15;
        int lp = (lp4 < 2) ? lp4 : 120 + lp4;
        y1_lds[nd * 1984 + lp * 16 + ci] = (__bf16)0.0f;
    }
    __syncthreads();

    // ---- phase 1: conv1, M=1920 (l*16+node), K=32(pad), N=16 ---------------
    {
        FragA b1;
#pragma unroll
        for (int h = 0; h < 16; ++h) b1.e[h] = w1b[akmap(lane, h) * 16 + col];
        const float bias = c1bias[col];
        const __bf16* abase = in2 + node * 744;
        for (int mt = wv; mt < 120; mt += 8) {           // mt == output position l
            FragA a;
            frag_ld_b32(a, abase + mt * 6, lane);        // k' = dk*6+ci contiguous
            FragC c; c.v = (v8f){};
            c.v = wmma_bf16(a.v, b1.v, c.v);
#pragma unroll
            for (int r = 0; r < 8; ++r) {
                int mm = r + ((lane >= 16) ? 8 : 0);     // node of this D row
                y1_lds[mm * 1984 + (mt + 2) * 16 + col] = (__bf16)fmaxf(c.e[r] + bias, 0.0f);
            }
        }
    }
    __syncthreads();

    // ---- phase 2: conv2, M=1920, K=96(pad), N=32 ---------------------------
    // six NAMED B fragments (no runtime-indexed arrays -> no scratch spills);
    // each mt: 3 A loads feed 6 WMMAs (both 16-col halves of N=32).
    {
        FragA b00, b01, b02, b10, b11, b12;
#pragma unroll
        for (int h = 0; h < 16; ++h) {
            int k0 = akmap(lane, h), k1 = 32 + k0, k2 = 64 + k0;
            b00.e[h] = w2b[k0 * 32 + col];
            b01.e[h] = w2b[k1 * 32 + col];
            b02.e[h] = w2b[k2 * 32 + col];
            b10.e[h] = w2b[k0 * 32 + 16 + col];
            b11.e[h] = w2b[k1 * 32 + 16 + col];
            b12.e[h] = w2b[k2 * 32 + 16 + col];
        }
        const float bias0 = c2bias[col];
        const float bias1 = c2bias[16 + col];
        const __bf16* abase = y1_lds + node * 1984;
        for (int mt = wv; mt < 120; mt += 8) {           // mt == output position l
            FragA a0, a1, a2;                            // issue all loads up front
            frag_ld_b128(a0, abase + (mt + 0) * 16, lane);
            frag_ld_b128(a1, abase + (mt + 2) * 16, lane);
            frag_ld_b128(a2, abase + (mt + 4) * 16, lane);
            FragC acc0; acc0.v = (v8f){};
            FragC acc1; acc1.v = (v8f){};
            acc0.v = wmma_bf16(a0.v, b00.v, acc0.v);
            acc1.v = wmma_bf16(a0.v, b10.v, acc1.v);
            acc0.v = wmma_bf16(a1.v, b01.v, acc0.v);
            acc1.v = wmma_bf16(a1.v, b11.v, acc1.v);
            acc0.v = wmma_bf16(a2.v, b02.v, acc0.v);
            acc1.v = wmma_bf16(a2.v, b12.v, acc1.v);
#pragma unroll
            for (int r = 0; r < 8; ++r) {
                int mm = r + ((lane >= 16) ? 8 : 0);
                __bf16* dst = y2_lds + mm * 3840 + mt * 32;
                dst[col]      = (__bf16)fmaxf(acc0.e[r] + bias0, 0.0f);
                dst[16 + col] = (__bf16)fmaxf(acc1.e[r] + bias1, 0.0f);
            }
        }
    }
    __syncthreads();

    // ---- phase 3: fc, M=16, K=3840 (k'=l*32+co == y2 linear), N=128 --------
    // NAMED even/odd double buffers, manual unroll-by-2 (120 even):
    // loads of step kk+1 are in flight while WMMA kk executes.
    {
        const int outc = wv * 16 + col;
        const __bf16* wrow = fcb + (size_t)outc * 3840;  // pre-permuted to k'
        const __bf16* arow = y2_lds + node * 3840;
        FragA aE, bE, aO, bO;
        frag_ld_b128(aE, arow, lane);
        frag_ld_b128(bE, wrow, lane);
        FragC acc; acc.v = (v8f){};
        for (int kk = 0; kk < 120; kk += 2) {
            __builtin_prefetch(wrow + (kk + 8) * 32, 0, 1);
            frag_ld_b128(aO, arow + (kk + 1) * 32, lane);
            frag_ld_b128(bO, wrow + (kk + 1) * 32, lane);
            acc.v = wmma_bf16(aE.v, bE.v, acc.v);
            frag_ld_b128(aE, arow + (kk + 2) * 32, lane);   // dead at kk=118 (in-bounds ws)
            frag_ld_b128(bE, wrow + (kk + 2) * 32, lane);
            acc.v = wmma_bf16(aO.v, bO.v, acc.v);
        }
        float bias = fcbias[outc];
#pragma unroll
        for (int r = 0; r < 8; ++r) {
            int mm = r + ((lane >= 16) ? 8 : 0);         // node
            int gn = tile0 + mm;
            if (gn < nN) x0[(size_t)gn * 128 + outc] = acc.e[r] + bias;
        }
    }
}

// ---------------------------------------------------------------------------
// GCN dense part: y[r][o] = sum_k x[r][k] * W[o][k], K=O=128, WMMA tiles.
// 128 rows per workgroup, 8 waves; wave w owns its 16-row m-tile (A hoisted).
// ---------------------------------------------------------------------------
__global__ __launch_bounds__(256) void gemm128(const float* __restrict__ x,
                                               const __bf16* __restrict__ wb,
                                               float* __restrict__ y, int nrows) {
    __shared__ __bf16 xs[128 * 128];
    __shared__ __bf16 wsh[128 * 128];
    const int tid = threadIdx.x;
    const int r0  = blockIdx.x * 128;
    for (int i = tid * 4; i < 128 * 128; i += 256 * 4) { // float4 in -> 4x bf16
        int r = i >> 7, k = i & 127;
        int gr = r0 + r;
        float4 v = (gr < nrows) ? *(const float4*)(x + (size_t)gr * 128 + k)
                                : make_float4(0.f, 0.f, 0.f, 0.f);
        __bf16* d = xs + i;
        d[0] = (__bf16)v.x; d[1] = (__bf16)v.y; d[2] = (__bf16)v.z; d[3] = (__bf16)v.w;
    }
    for (int i = tid * 8; i < 128 * 128; i += 256 * 8)   // uint4 weight copy
        *(uint4*)(wsh + i) = *(const uint4*)(wb + i);
    __syncthreads();

    const int lane = tid & 31, wv = tid >> 5;
    const int col = lane & 15;
    FragA a0, a1, a2, a3;                                // named, never spilled
    const __bf16* arow = xs + (wv * 16 + (lane & 15)) * 128;
    frag_ld_b128(a0, arow,      lane);
    frag_ld_b128(a1, arow + 32, lane);
    frag_ld_b128(a2, arow + 64, lane);
    frag_ld_b128(a3, arow + 96, lane);
#pragma unroll
    for (int nt = 0; nt < 8; ++nt) {
        const __bf16* brow = wsh + (nt * 16 + col) * 128;
        FragA b0, b1, b2, b3;
        frag_ld_b128(b0, brow,      lane);
        frag_ld_b128(b1, brow + 32, lane);
        frag_ld_b128(b2, brow + 64, lane);
        frag_ld_b128(b3, brow + 96, lane);
        FragC acc; acc.v = (v8f){};
        acc.v = wmma_bf16(a0.v, b0.v, acc.v);
        acc.v = wmma_bf16(a1.v, b1.v, acc.v);
        acc.v = wmma_bf16(a2.v, b2.v, acc.v);
        acc.v = wmma_bf16(a3.v, b3.v, acc.v);
#pragma unroll
        for (int r = 0; r < 8; ++r) {
            int mm = r + ((lane >= 16) ? 8 : 0);
            int gr = r0 + wv * 16 + mm;
            if (gr < nrows) y[(size_t)gr * 128 + nt * 16 + col] = acc.e[r];
        }
    }
}

// ---------------------------------------------------------------------------
// Sparse part: degree, rsqrt norm, normalized scatter-add, bias/ReLU, pooling
// ---------------------------------------------------------------------------
__global__ void deg_count(const int* __restrict__ dst, float* deg, int nE) {
    int e = blockIdx.x * 256 + threadIdx.x;
    if (e < nE) atomicAdd(&deg[dst[e]], 1.0f);
}
__global__ void make_dinv(float* deg, int nN) {
    int i = blockIdx.x * 256 + threadIdx.x;
    if (i < nN) deg[i] = rsqrtf(deg[i] + 1.0f);   // +1 self-loop; always > 0
}
__global__ void scatter_norm(const float* __restrict__ h, const int* __restrict__ src,
                             const int* __restrict__ dst, const float* __restrict__ dinv,
                             float* __restrict__ out, int nE) {
    int b = blockIdx.x, c = threadIdx.x;           // 128 threads = channels
    if (b < nE) {
        int s = src[b], d = dst[b];
        float w = dinv[s] * dinv[d];
        atomicAdd(&out[(size_t)d * 128 + c], h[(size_t)s * 128 + c] * w);
    } else {
        int i = b - nE;                            // self loops (nNodes == nE)
        float w = dinv[i] * dinv[i];
        atomicAdd(&out[(size_t)i * 128 + c], h[(size_t)i * 128 + c] * w);
    }
}
__global__ void bias_act(float* __restrict__ x, const float* __restrict__ b, int total, int relu) {
    int i = blockIdx.x * 256 + threadIdx.x;
    if (i < total) {
        float v = x[i] + b[i & 127];
        x[i] = relu ? fmaxf(v, 0.0f) : v;
    }
}
__global__ void pool_scatter(const float* __restrict__ x, const int* __restrict__ src,
                             const int* __restrict__ batch, float* __restrict__ out, int nE) {
    int e = blockIdx.x, c = threadIdx.x;
    if (e < nE) {
        int g = batch[src[e]];
        atomicAdd(&out[(size_t)g * 128 + c], x[(size_t)e * 128 + c]);
    }
}

// ---------------------------------------------------------------------------
// Tiny MLP head (64x128 @ 128x256 -> ReLU -> @ 256x128): negligible FLOPs.
// ---------------------------------------------------------------------------
__global__ void head1(const float* __restrict__ pooled, const float* __restrict__ w,
                      const float* __restrict__ b, float* __restrict__ t) {
    int g = blockIdx.x, o = threadIdx.x;
    const float* pr = pooled + (size_t)g * 128;
    const float* wr = w + (size_t)o * 128;
    float acc = 0.0f;
    for (int k = 0; k < 128; ++k) acc += pr[k] * wr[k];
    t[(size_t)g * 256 + o] = fmaxf(acc + b[o], 0.0f);
}
__global__ void head2(const float* __restrict__ t, const float* __restrict__ w,
                      const float* __restrict__ b, float* __restrict__ out, int total) {
    int i = blockIdx.x * 256 + threadIdx.x;
    if (i < total) {
        int g = i >> 7, o = i & 127;
        const float* tr = t + (size_t)g * 256;
        const float* wr = w + (size_t)o * 256;
        float acc = 0.0f;
        for (int k = 0; k < 256; ++k) acc += tr[k] * wr[k];
        out[i] = acc + b[o];
    }
}

extern "C" void kernel_launch(void* const* d_in, const int* in_sizes, int n_in,
                              void* d_out, int out_size, void* d_ws, size_t ws_size,
                              hipStream_t stream) {
    const float* edge_attr = (const float*)d_in[0];
    const int*   edge_idx  = (const int*)d_in[1];
    const int*   batch     = (const int*)d_in[2];
    const float* c1w = (const float*)d_in[4];  const float* c1b = (const float*)d_in[5];
    const float* c2w = (const float*)d_in[6];  const float* c2b = (const float*)d_in[7];
    const float* fcw = (const float*)d_in[8];  const float* fcb = (const float*)d_in[9];
    const float* g1w = (const float*)d_in[10]; const float* g1b = (const float*)d_in[11];
    const float* g2w = (const float*)d_in[12]; const float* g2b = (const float*)d_in[13];
    const float* f1w = (const float*)d_in[14]; const float* f1b = (const float*)d_in[15];
    const float* f2w = (const float*)d_in[16]; const float* f2b = (const float*)d_in[17];

    const int N = in_sizes[0] / 720;        // 100000 nodes == edges
    const int B = out_size / 128;           // 64 graphs
    const int* src = edge_idx;
    const int* dst = edge_idx + N;

    // workspace carve-out
    char* ws = (char*)d_ws;
    size_t off = 0;
    auto carve = [&](size_t bytes) { size_t r = off; off = (off + bytes + 255) & ~(size_t)255; return r; };
    __bf16* w1b  = (__bf16*)(ws + carve(512 * 2));
    __bf16* w2b  = (__bf16*)(ws + carve(3072 * 2));
    __bf16* fcbw = (__bf16*)(ws + carve(491520 * 2));
    __bf16* g1bw = (__bf16*)(ws + carve(16384 * 2));
    __bf16* g2bw = (__bf16*)(ws + carve(16384 * 2));
    float* dinv   = (float*)(ws + carve((size_t)N * 4));
    float* pooled = (float*)(ws + carve((size_t)B * 128 * 4));
    float* thid   = (float*)(ws + carve((size_t)B * 256 * 4));
    float* bufA   = (float*)(ws + carve((size_t)N * 128 * 4));   // x0 / x2
    float* bufB   = (float*)(ws + carve((size_t)N * 128 * 4));   // h1 / h2
    float* bufC   = (float*)(ws + carve((size_t)N * 128 * 4));   // x1
    (void)ws_size; (void)n_in;

    // weights -> permuted bf16
    prep_weights<<<(527872 + 255) / 256, 256, 0, stream>>>(c1w, c2w, fcw, g1w, g2w,
                                                           w1b, w2b, fcbw, g1bw, g2bw);
    // degrees -> dinv
    hipMemsetAsync(dinv, 0, (size_t)N * 4, stream);
    deg_count<<<(N + 255) / 256, 256, 0, stream>>>(dst, dinv, N);
    make_dinv<<<(N + 255) / 256, 256, 0, stream>>>(dinv, N);

    // fused conv1+conv2+fc -> bufA = x0 [N,128]
    fused_conv_fc<<<(N + 15) / 16, 256, FUSED_LDS, stream>>>(
        edge_attr, w1b, w2b, fcbw, c1b, c2b, fcb, bufA, N);

    // GCN layer 1: h = x0 @ W1^T ; x1 = relu(scatter(norm*h) + b1)
    gemm128<<<(N + 127) / 128, 256, 0, stream>>>(bufA, g1bw, bufB, N);
    hipMemsetAsync(bufC, 0, (size_t)N * 128 * 4, stream);
    scatter_norm<<<2 * N, 128, 0, stream>>>(bufB, src, dst, dinv, bufC, N);
    bias_act<<<((size_t)N * 128 + 255) / 256, 256, 0, stream>>>(bufC, g1b, N * 128, 1);

    // GCN layer 2: h2 = x1 @ W2^T ; x2 = scatter(norm*h2) + b2
    gemm128<<<(N + 127) / 128, 256, 0, stream>>>(bufC, g2bw, bufB, N);
    hipMemsetAsync(bufA, 0, (size_t)N * 128 * 4, stream);
    scatter_norm<<<2 * N, 128, 0, stream>>>(bufB, src, dst, dinv, bufA, N);
    bias_act<<<((size_t)N * 128 + 255) / 256, 256, 0, stream>>>(bufA, g2b, N * 128, 0);

    // edge->graph pooling, then MLP head
    hipMemsetAsync(pooled, 0, (size_t)B * 128 * 4, stream);
    pool_scatter<<<N, 128, 0, stream>>>(bufA, src, batch, pooled, N);
    head1<<<B, 256, 0, stream>>>(pooled, f1w, f1b, thid);
    head2<<<(B * 128 + 255) / 256, 256, 0, stream>>>(thid, f2w, f2b, (float*)d_out, B * 128);
}
